// NeuralHMM_12927851561467
// MI455X (gfx1250) — matrix operand order repaired and checked by reading, not compiled, not run on hardware
//
#include <hip/hip_runtime.h>
#include <hip/hip_bf16.h>
#include <math.h>

// ---------------------------------------------------------------------------
// NeuralHMM on MI455X (gfx1250): bf16 WMMA GEMMs + fused online log-softmax
// target-gather + tiny HMM forward scan.
// ---------------------------------------------------------------------------

#define HMM_V    2048
#define HMM_K    16
#define HMM_EMB  128
#define HMM_HID  1024
#define HMM_CTX  5            // 2C+1
#define HMM_IND  (HMM_EMB * HMM_CTX)   // 640
#define HMM_B    4
#define HMM_T    512
#define HMM_BT   (HMM_B * HMM_T)       // 2048

typedef __attribute__((ext_vector_type(16))) __bf16          v16bf;
typedef __attribute__((ext_vector_type(8)))  float           v8f;
typedef __attribute__((ext_vector_type(8)))  unsigned short  v8u16;
typedef __attribute__((ext_vector_type(16))) unsigned short  v16u16;

__device__ __forceinline__ unsigned short f2bf(float f) {
    unsigned int u = __float_as_uint(f);
    u += 0x7FFFu + ((u >> 16) & 1u);           // round-to-nearest-even
    return (unsigned short)(u >> 16);
}

__device__ __forceinline__ v16bf join16(v8u16 lo, v8u16 hi) {
    v16u16 u;
#pragma unroll
    for (int i = 0; i < 8; ++i) { u[i] = lo[i]; u[i + 8] = hi[i]; }
    return __builtin_bit_cast(v16bf, u);
}

// ---------------------------------------------------------------------------
// Prep kernels: gather windowed context embeddings (bf16), fp32->bf16 weights
// ---------------------------------------------------------------------------
__global__ void hmm_gather_E(const int* __restrict__ x,
                             const float* __restrict__ emb,
                             unsigned short* __restrict__ E) {
    int i = blockIdx.x * blockDim.x + threadIdx.x;     // over BT*640
    if (i >= HMM_BT * HMM_IND) return;
    int pos = i / HMM_IND, col = i - pos * HMM_IND;
    int j = col >> 7, e = col & 127;                   // window slot, emb elem
    int b = pos >> 9, t = pos & (HMM_T - 1);
    int tt = t + j - 2;                                // C = 2
    int tok = (tt < 0 || tt >= HMM_T) ? 0 : x[b * HMM_T + tt];  // PAD = 0
    E[i] = f2bf(emb[tok * HMM_EMB + e]);
}

__global__ void hmm_cvt_bf16(const float* __restrict__ src,
                             unsigned short* __restrict__ dst, int n) {
    int i = blockIdx.x * blockDim.x + threadIdx.x;
    if (i < n) dst[i] = f2bf(src[i]);
}

// ---------------------------------------------------------------------------
// GEMM1: h = relu(E (2048x640) * w1^T (640x1024) + b1), bf16 WMMA, f32 acc.
// 4 waves/WG; each wave owns one 16x16 output tile.
// ---------------------------------------------------------------------------
__global__ void __launch_bounds__(128)
hmm_gemm1_relu(const unsigned short* __restrict__ E,
               const unsigned short* __restrict__ w1b,
               const float* __restrict__ b1,
               unsigned short* __restrict__ h) {
    const int tid  = threadIdx.x;
    const int wave = tid >> 5, lane = tid & 31;
    const int half = lane >> 4, l16 = lane & 15;
    const int mBase = blockIdx.x * 16;
    const int n     = blockIdx.y * 64 + wave * 16 + l16;

    const unsigned short* arow = E   + (size_t)(mBase + l16) * HMM_IND;
    const unsigned short* brow = w1b + (size_t)n * HMM_IND + half * 16;

    v8f c = {};
#pragma unroll 4
    for (int kb = 0; kb < HMM_IND; kb += 32) {
        v16bf a = join16(*(const v8u16*)(arow + kb + half * 16),
                         *(const v8u16*)(arow + kb + half * 16 + 8));
        v16bf b = join16(*(const v8u16*)(brow + kb),
                         *(const v8u16*)(brow + kb + 8));
        c = __builtin_amdgcn_wmma_f32_16x16x32_bf16(false, a, false, b,
                                                    (short)0, c, false, false);
    }
    const float bias = b1[n];
#pragma unroll
    for (int r = 0; r < 8; ++r) {
        float v = fmaxf(c[r] + bias, 0.0f);
        int m = mBase + r + 8 * half;
        h[(size_t)m * HMM_HID + n] = f2bf(v);
    }
}

// ---------------------------------------------------------------------------
// GEMM2 + fused online log-softmax + target gather.
// Block: 512 threads = 16 waves = 4 M-tiles x 4 column strips.
// Covers 64 positions x one HMM state k (V=2048 logit columns, never stored).
// Each wave computes 16 rows x 64 cols per iteration (4 accumulators), so a
// single A-operand LDS load feeds 4 WMMAs. h tile (64x1024 bf16 = 128 KB)
// staged in LDS (WGP has 320 KB). w2(bf16, 64 MB) is L2-resident (192 MB L2);
// next strip is prefetched (global_prefetch_b8) while current one computes.
// ---------------------------------------------------------------------------
__global__ void __launch_bounds__(512)
hmm_gemm2_logsm(const unsigned short* __restrict__ h,
                const unsigned short* __restrict__ w2b,
                const float* __restrict__ b2,
                const int* __restrict__ x,
                float* __restrict__ log_e) {
    extern __shared__ char smem[];
    unsigned short* hsh = (unsigned short*)smem;              // 64*1024 bf16
    float* mergeM = (float*)(smem + 64 * HMM_HID * 2);        // [4][64]
    float* mergeS = mergeM + 256;
    float* mergeT = mergeS + 256;

    const int posBase = blockIdx.x * 64;
    const int k   = blockIdx.y;
    const int tid = threadIdx.x;
    const int wave = tid >> 5, lane = tid & 31;
    const int mt = wave >> 2, cs = wave & 3;
    const int half = lane >> 4, l16 = lane & 15;

    {   // cooperative load of 64x1024 bf16 h tile (16B per thread per iter)
        const uint4* src = (const uint4*)(h + (size_t)posBase * HMM_HID);
        uint4* dst = (uint4*)hsh;
        for (int i = tid; i < 64 * HMM_HID / 8; i += 512) dst[i] = src[i];
    }
    __syncthreads();

    int xr[8];
#pragma unroll
    for (int r = 0; r < 8; ++r)
        xr[r] = x[posBase + mt * 16 + r + 8 * half];

    float M[8], S[8], Tg[8];
#pragma unroll
    for (int r = 0; r < 8; ++r) { M[r] = -1e30f; S[r] = 0.0f; Tg[r] = -1e30f; }

    const unsigned short* hrow = hsh + (size_t)(mt * 16 + l16) * HMM_HID;

    for (int it = 0; it < 8; ++it) {
        const int colBase = cs * 64 + it * 256;          // columns within state
        const int orow0   = k * HMM_V + colBase + l16;   // w2 row = logit col
        const unsigned short* brow[4];
#pragma unroll
        for (int t = 0; t < 4; ++t)
            brow[t] = w2b + (size_t)(orow0 + 16 * t) * HMM_HID + half * 16;

        if (it < 7) {   // prefetch next 256-col strip toward the WGP
#pragma unroll
            for (int t = 0; t < 4; ++t)
                __builtin_prefetch(brow[t] + (size_t)256 * HMM_HID, 0, 0);
        }

        v8f acc[4] = {};
#pragma unroll 2
        for (int kb = 0; kb < HMM_HID; kb += 32) {
            v16bf a = join16(*(const v8u16*)(hrow + kb + half * 16),
                             *(const v8u16*)(hrow + kb + half * 16 + 8));
#pragma unroll
            for (int t = 0; t < 4; ++t) {
                v16bf b = join16(*(const v8u16*)(brow[t] + kb),
                                 *(const v8u16*)(brow[t] + kb + 8));
                acc[t] = __builtin_amdgcn_wmma_f32_16x16x32_bf16(
                             false, a, false, b, (short)0, acc[t], false, false);
            }
        }
        float bias[4];
#pragma unroll
        for (int t = 0; t < 4; ++t) bias[t] = b2[orow0 + 16 * t];
        const int ncol = colBase + l16;
#pragma unroll
        for (int r = 0; r < 8; ++r) {
            float v0 = acc[0][r] + bias[0];
            float v1 = acc[1][r] + bias[1];
            float v2 = acc[2][r] + bias[2];
            float v3 = acc[3][r] + bias[3];
            if (ncol      == xr[r]) Tg[r] = v0;
            if (ncol + 16 == xr[r]) Tg[r] = v1;
            if (ncol + 32 == xr[r]) Tg[r] = v2;
            if (ncol + 48 == xr[r]) Tg[r] = v3;
            float mx = fmaxf(fmaxf(v0, v1), fmaxf(v2, v3));
#pragma unroll
            for (int off = 1; off < 16; off <<= 1)
                mx = fmaxf(mx, __shfl_xor(mx, off, 32));
            float se = expf(v0 - mx) + expf(v1 - mx) +
                       expf(v2 - mx) + expf(v3 - mx);
#pragma unroll
            for (int off = 1; off < 16; off <<= 1)
                se += __shfl_xor(se, off, 32);
            float nm = fmaxf(M[r], mx);       // online merge
            S[r] = S[r] * expf(M[r] - nm) + se * expf(mx - nm);
            M[r] = nm;
        }
    }
#pragma unroll
    for (int r = 0; r < 8; ++r) {             // broadcast captured target logit
        float t = Tg[r];
#pragma unroll
        for (int off = 1; off < 16; off <<= 1)
            t = fmaxf(t, __shfl_xor(t, off, 32));
        Tg[r] = t;
    }
    if (l16 == 0) {
#pragma unroll
        for (int r = 0; r < 8; ++r) {
            int row = mt * 16 + r + 8 * half;
            mergeM[cs * 64 + row] = M[r];
            mergeS[cs * 64 + row] = S[r];
            mergeT[cs * 64 + row] = Tg[r];
        }
    }
    __syncthreads();
    if (tid < 64) {                           // merge 4 column strips per row
        float m = -1e30f, s = 0.0f, tg = -1e30f;
#pragma unroll
        for (int c2 = 0; c2 < 4; ++c2) {
            float mm = mergeM[c2 * 64 + tid];
            float ss = mergeS[c2 * 64 + tid];
            float nm = fmaxf(m, mm);
            s = s * expf(m - nm) + ss * expf(mm - nm);
            m = nm;
            tg = fmaxf(tg, mergeT[c2 * 64 + tid]);
        }
        log_e[(size_t)(posBase + tid) * HMM_K + k] = tg - (m + logf(s));
    }
}

// ---------------------------------------------------------------------------
// HMM forward recursion: B=4, K=16 -> 64 threads, one block, sequential in T.
// ---------------------------------------------------------------------------
__global__ void __launch_bounds__(64)
hmm_forward(const float* __restrict__ log_e,
            const int* __restrict__ lengths,
            const float* __restrict__ start_logits,
            const float* __restrict__ trans_logits,
            float* __restrict__ out) {
    __shared__ float alpha[HMM_B][HMM_K];
    __shared__ float lt[HMM_K][HMM_K];
    __shared__ float ls[HMM_K];
    __shared__ float ltl[HMM_K];
    const int tid = threadIdx.x;
    const int b = tid >> 4, kk = tid & 15;

    if (tid < 16) {
        float m = -1e30f;
        for (int j = 0; j < 16; ++j) m = fmaxf(m, trans_logits[tid * 16 + j]);
        float s = 0.0f;
        for (int j = 0; j < 16; ++j) s += expf(trans_logits[tid * 16 + j] - m);
        ltl[tid] = m + logf(s);
        float m0 = -1e30f;
        for (int j = 0; j < 16; ++j) m0 = fmaxf(m0, start_logits[j]);
        float s0 = 0.0f;
        for (int j = 0; j < 16; ++j) s0 += expf(start_logits[j] - m0);
        ls[tid] = start_logits[tid] - (m0 + logf(s0));
    }
    __syncthreads();
    for (int i = tid; i < 256; i += 64)
        lt[i >> 4][i & 15] = trans_logits[i] - ltl[i >> 4];
    __syncthreads();

    alpha[b][kk] = ls[kk] + log_e[(size_t)(b * HMM_T) * HMM_K + kk];
    __syncthreads();

    const int len = lengths[b];
    for (int t = 1; t < HMM_T; ++t) {
        float m = -1e30f;
#pragma unroll
        for (int j = 0; j < 16; ++j) m = fmaxf(m, alpha[b][j] + lt[j][kk]);
        float s = 0.0f;
#pragma unroll
        for (int j = 0; j < 16; ++j) s += expf(alpha[b][j] + lt[j][kk] - m);
        float val = log_e[(size_t)(b * HMM_T + t) * HMM_K + kk] + m + logf(s);
        __syncthreads();
        if (t < len) alpha[b][kk] = val;
        __syncthreads();
    }
    if (kk == 0) {
        float m = -1e30f;
        for (int j = 0; j < 16; ++j) m = fmaxf(m, alpha[b][j]);
        float s = 0.0f;
        for (int j = 0; j < 16; ++j) s += expf(alpha[b][j] - m);
        out[b] = (len > 0) ? (m + logf(s)) : 0.0f;
    }
}

// ---------------------------------------------------------------------------
extern "C" void kernel_launch(void* const* d_in, const int* in_sizes, int n_in,
                              void* d_out, int out_size, void* d_ws, size_t ws_size,
                              hipStream_t stream) {
    const int*   x       = (const int*)  d_in[0];
    const int*   lengths = (const int*)  d_in[1];
    const float* emb     = (const float*)d_in[2];
    const float* w1      = (const float*)d_in[3];
    const float* b1      = (const float*)d_in[4];
    const float* w2      = (const float*)d_in[5];
    const float* b2      = (const float*)d_in[6];
    const float* startl  = (const float*)d_in[7];
    const float* transl  = (const float*)d_in[8];
    float*       out     = (float*)d_out;

    size_t off = 0;
    auto alloc = [&](size_t bytes) -> void* {
        void* p = (char*)d_ws + off;
        off += (bytes + 255) & ~(size_t)255;
        return p;
    };
    unsigned short* E    = (unsigned short*)alloc((size_t)HMM_BT * HMM_IND * 2);
    unsigned short* w1b  = (unsigned short*)alloc((size_t)HMM_HID * HMM_IND * 2);
    unsigned short* w2b  = (unsigned short*)alloc((size_t)HMM_K * HMM_V * HMM_HID * 2);
    unsigned short* hbf  = (unsigned short*)alloc((size_t)HMM_BT * HMM_HID * 2);
    float*          loge = (float*)alloc((size_t)HMM_BT * HMM_K * 4);

    // 1) gather context embeddings -> bf16
    {
        int n = HMM_BT * HMM_IND;
        hmm_gather_E<<<(n + 255) / 256, 256, 0, stream>>>(x, emb, E);
    }
    // 2) weights fp32 -> bf16 (w2 becomes 64 MB: L2-resident on 192 MB L2)
    {
        int n1 = HMM_HID * HMM_IND;
        hmm_cvt_bf16<<<(n1 + 255) / 256, 256, 0, stream>>>(w1, w1b, n1);
        int n2 = HMM_K * HMM_V * HMM_HID;
        hmm_cvt_bf16<<<(n2 + 255) / 256, 256, 0, stream>>>(w2, w2b, n2);
    }
    // 3) GEMM1 + ReLU
    hmm_gemm1_relu<<<dim3(HMM_BT / 16, HMM_HID / 64), 128, 0, stream>>>(E, w1b, b1, hbf);
    // 4) GEMM2 + fused online log-softmax + target gather
    {
        size_t shmem = (size_t)64 * HMM_HID * 2 + 3 * 256 * sizeof(float);
        hmm_gemm2_logsm<<<dim3(HMM_BT / 64, HMM_K), 512, shmem, stream>>>(
            hbf, w2b, b2, x, loge);
    }
    // 5) HMM forward scan
    hmm_forward<<<1, 64, 0, stream>>>(loge, lengths, startl, transl, out);
}
